// CrossShareUnit_89111981457827
// MI455X (gfx1250) — matrix operand order repaired and checked by reading, not compiled, not run on hardware
//
#include <hip/hip_runtime.h>
#include <hip/hip_bf16.h>

typedef __attribute__((ext_vector_type(16))) __bf16 v16bf;
typedef __attribute__((ext_vector_type(8)))  float  v8f;

#define B_TOT 8192
#define DHID  1024
#define KDIM  8

__device__ __forceinline__ unsigned short f2bf(float f){
  unsigned int u = __float_as_uint(f);
  unsigned int r = u + 0x7fffu + ((u >> 16) & 1u);
  return (unsigned short)(r >> 16);
}
__device__ __forceinline__ float bf2f(unsigned short h){
  return __uint_as_float(((unsigned int)h) << 16);
}

union Frag16 { uint4 q[2]; v16bf v; };

// ---------------------------------------------------------------------------
// Pack a logical B-matrix B[d,n] = src[src_off + d*sd + n*sn] (f32) into
// bf16 WMMA B-fragments. Fragment f = n_tile*ndch + d_chunk, 512 bf16 each;
// lane l owns 16 contiguous bf16: rows d = d_chunk*32 + (l>>4)*16 + i,
// column n = n_tile*16 + (l&15).  8 fragments per 256-thread block.
// ---------------------------------------------------------------------------
__global__ void pack_b_kernel(const float* __restrict__ src, long src_off,
                              long sd, long sn, int ndch,
                              unsigned short* __restrict__ dst)
{
  const int f    = blockIdx.x * 8 + (threadIdx.x >> 5);
  const int lane = threadIdx.x & 31;
  const int n_tile = f / ndch;
  const int dch    = f % ndch;
  long base = src_off + (long)(dch * 32 + (lane >> 4) * 16) * sd
                      + (long)(n_tile * 16 + (lane & 15)) * sn;
  unsigned short* d = dst + (size_t)f * 512 + lane * 16;
  #pragma unroll
  for (int i = 0; i < 16; ++i) d[i] = f2bf(src[base + (long)i * sd]);
}

// ---------------------------------------------------------------------------
// g_vec = (h @ W1^T + b1) @ W2^T + b2  for a 64-row batch tile. All GEMM work
// on v_wmma_f32_16x16x32_bf16. 8 waves: 4 M-subtiles x 2 N-halves.
// Weight fragments addressed as uniform-base + 32-bit lane offset + immediate.
// ---------------------------------------------------------------------------
__global__ __launch_bounds__(256) void mlp_gvec_kernel(
    const float* __restrict__ h, const char* __restrict__ w1c,
    const float* __restrict__ b1, const char* __restrict__ w2c,
    const float* __restrict__ b2, float* __restrict__ gvec, int N1)
{
  extern __shared__ char dynsm[];
  unsigned short* hsm = (unsigned short*)dynsm;  // 64 x 1024 bf16
  unsigned short* hid = hsm + 64 * 1024;         // 64 x N1   bf16

  const int b0  = blockIdx.x * 64;
  const int tid = threadIdx.x;

  { // stage h tile -> bf16 LDS
    const float4* s4 = (const float4*)(h + (size_t)b0 * DHID);
    ushort4* d4 = (ushort4*)hsm;
    #pragma unroll
    for (int i = 0; i < 64; ++i) {
      int idx = tid + 256 * i;
      float4 v = s4[idx];
      ushort4 o; o.x=f2bf(v.x); o.y=f2bf(v.y); o.z=f2bf(v.z); o.w=f2bf(v.w);
      d4[idx] = o;
    }
  }
  __syncthreads();

  const int wave = tid >> 5, lane = tid & 31;
  const int msub = wave & 3;          // 16-row M-subtile
  const int nsel = wave >> 2;         // N-range half
  const int laneN = lane & 15, laneHi = lane >> 4;
  const int laneByte = lane * 32;
  const int arow   = msub * 16 + laneN;
  const int rowSub = msub * 16 + laneHi * 8;
  const uint4* hq = (const uint4*)hsm;

  // stage 1: hid = h @ W1^T + b1
  const int half1 = (N1 / 16) / 2;
  for (int ntl = 0; ntl < half1; ++ntl) {
    const int nt = nsel * half1 + ntl;
    const char* bp = w1c + (unsigned)(nt * 32 * 1024) + laneByte;
    v8f acc = {};
    #pragma unroll 8
    for (int dch = 0; dch < 32; ++dch) {
      Frag16 a, b;
      int off = arow * 128 + dch * 4 + laneHi * 2;
      a.q[0] = hq[off]; a.q[1] = hq[off + 1];
      b.q[0] = *(const uint4*)(bp + dch * 1024);
      b.q[1] = *(const uint4*)(bp + dch * 1024 + 16);
      acc = __builtin_amdgcn_wmma_f32_16x16x32_bf16(
              false, a.v, false, b.v, (short)0, acc, false, false);
    }
    const int n = nt * 16 + laneN;
    const float bias = b1[n];
    #pragma unroll
    for (int r = 0; r < 8; ++r)
      hid[(rowSub + r) * N1 + n] = f2bf(acc[r] + bias);
  }
  __syncthreads();

  // stage 2: gvec = hid @ W2^T + b2
  const int dch2 = N1 / 32;
  const int rsq  = N1 >> 3;           // hid row stride in uint4
  const uint4* hidq = (const uint4*)hid;
  for (int ntl = 0; ntl < 32; ++ntl) {
    const int nt = nsel * 32 + ntl;
    const char* bp = w2c + (unsigned)(nt * dch2 * 1024) + laneByte;
    v8f acc = {};
    #pragma unroll
    for (int dch = 0; dch < dch2; ++dch) {
      Frag16 a, b;
      int off = arow * rsq + dch * 4 + laneHi * 2;
      a.q[0] = hidq[off]; a.q[1] = hidq[off + 1];
      b.q[0] = *(const uint4*)(bp + dch * 1024);
      b.q[1] = *(const uint4*)(bp + dch * 1024 + 16);
      acc = __builtin_amdgcn_wmma_f32_16x16x32_bf16(
              false, a.v, false, b.v, (short)0, acc, false, false);
    }
    const int n = nt * 16 + laneN;
    const float bias = b2[n];
    #pragma unroll
    for (int r = 0; r < 8; ++r)
      gvec[(size_t)(b0 + rowSub + r) * DHID + n] = acc[r] + bias;
  }
}

// ---------------------------------------------------------------------------
// t[b,k] = h_a[b,:] . G[:,k,:] . h_b[b,:]^T  (pre-tanh) for a 32-row tile.
// Loop nest: d-slice (2 x 512-wide, A-fragments cached in 128 VGPRs)
//   -> e-tile (h_b values hoisted, shared across both k)
//     -> k (2 per wave) -> 16 chained WMMAs with immediate-offset B loads.
// Per-WMMA steady state: 2 global_load_b128 (imm offsets off one 32-bit
// address), 0.25 ds loads, ~0 address VALU.
// ---------------------------------------------------------------------------
__global__ __launch_bounds__(256) void cross_bilinear_kernel(
    const float* __restrict__ lh, const float* __restrict__ mh,
    const char* __restrict__ gpack, float* __restrict__ tOut)
{
  extern __shared__ char dynsm[];
  unsigned short* ha_bf = (unsigned short*)dynsm;   // 32 x 1024
  unsigned short* hb_bf = ha_bf + 32 * 1024;        // 32 x 1024

  const int dir = blockIdx.y;
  const float* hA = dir ? mh : lh;
  const float* hB = dir ? lh : mh;
  const char* gbase = gpack + (size_t)dir * 16777216u;   // 16 MB packed G / dir
  float* tD = tOut + (size_t)dir * B_TOT * KDIM;

  const int b0  = blockIdx.x * 32;
  const int tid = threadIdx.x;

  { // stage both operand tiles -> bf16 LDS
    const float4* a4 = (const float4*)(hA + (size_t)b0 * DHID);
    const float4* b4 = (const float4*)(hB + (size_t)b0 * DHID);
    ushort4* da = (ushort4*)ha_bf;
    ushort4* db = (ushort4*)hb_bf;
    #pragma unroll
    for (int i = 0; i < 32; ++i) {
      int idx = tid + 256 * i;
      float4 va = a4[idx]; float4 vb = b4[idx];
      ushort4 oa, ob;
      oa.x=f2bf(va.x); oa.y=f2bf(va.y); oa.z=f2bf(va.z); oa.w=f2bf(va.w);
      ob.x=f2bf(vb.x); ob.y=f2bf(vb.y); ob.z=f2bf(vb.z); ob.w=f2bf(vb.w);
      da[idx] = oa; db[idx] = ob;
    }
  }
  __syncthreads();

  const int wave = tid >> 5, lane = tid & 31;
  const int mb = wave & 1;            // 16-row M-subtile
  const int kp = wave >> 1;           // pair of k values
  const int laneN = lane & 15, laneHi = lane >> 4;
  const int laneByte = lane * 32;
  const int rowSub = mb * 16 + laneHi * 8;   // C-layout row base
  const int arow   = mb * 16 + laneN;

  float tacc[2][8];
  #pragma unroll
  for (int kk = 0; kk < 2; ++kk)
    #pragma unroll
    for (int r = 0; r < 8; ++r) tacc[kk][r] = 0.f;

  const uint4* haq = (const uint4*)ha_bf;

  for (int d8 = 0; d8 < 2; ++d8) {    // 512-wide slices of the 1024 d-dim
    Frag16 afr[16];                   // cache A fragments (128 VGPRs)
    #pragma unroll
    for (int j = 0; j < 16; ++j) {
      int off = arow * 128 + d8 * 64 + j * 4 + laneHi * 2;  // uint4 units
      afr[j].q[0] = haq[off];
      afr[j].q[1] = haq[off + 1];
    }
    for (int et = 0; et < 64; ++et) {
      // h_b values for this e-column, shared by both k of this wave
      float hbv[8];
      const unsigned short* hbp = hb_bf + rowSub * DHID + et * 16 + laneN;
      #pragma unroll
      for (int r = 0; r < 8; ++r) hbv[r] = bf2f(hbp[r * DHID]);

      #pragma unroll
      for (int kk = 0; kk < 2; ++kk) {
        const int k = kp * 2 + kk;
        const unsigned voff =
            (unsigned)(((k * 64 + et) * 32 + d8 * 16) * 1024) + laneByte;
        const char* bp = gbase + voff;
        __builtin_prefetch(bp + 16 * 1024, 0, 0);   // warm next d-slice
        v8f acc = {};
        #pragma unroll
        for (int j = 0; j < 16; ++j) {
          Frag16 bfr;
          bfr.q[0] = *(const uint4*)(bp + j * 1024);
          bfr.q[1] = *(const uint4*)(bp + j * 1024 + 16);
          acc = __builtin_amdgcn_wmma_f32_16x16x32_bf16(
                  false, afr[j].v, false, bfr.v, (short)0, acc, false, false);
        }
        #pragma unroll
        for (int r = 0; r < 8; ++r)
          tacc[kk][r] = fmaf(acc[r], hbv[r], tacc[kk][r]);
      }
    }
  }

  // reduce the 16 e-columns held across the half-wave, one writer per (b,k)
  #pragma unroll
  for (int kk = 0; kk < 2; ++kk) {
    #pragma unroll
    for (int r = 0; r < 8; ++r) {
      float s = tacc[kk][r];
      s += __shfl_xor(s, 1, 32);
      s += __shfl_xor(s, 2, 32);
      s += __shfl_xor(s, 4, 32);
      s += __shfl_xor(s, 8, 32);
      if (laneN == 0)
        tD[(size_t)(b0 + rowSub + r) * KDIM + (kp * 2 + kk)] = s;
    }
  }
}

// ---------------------------------------------------------------------------
// S = sum_k tanh(t[b,k]); attn = softmax(S * g_vec[b,:]); out = h_a*(1+attn).
// One wave per batch row, 1024 elements held in 32 regs/lane.
// ---------------------------------------------------------------------------
__global__ __launch_bounds__(256) void softmax_out_kernel(
    const float* __restrict__ lh, const float* __restrict__ mh,
    const float* __restrict__ gvec, const float* __restrict__ tIn,
    float* __restrict__ out)
{
  const int dir = blockIdx.y;
  const float* hA = dir ? mh : lh;
  const float* gv = gvec + (size_t)dir * B_TOT * DHID;
  const float* tD = tIn + (size_t)dir * B_TOT * KDIM;
  float* outMain = out + (size_t)dir * B_TOT * DHID;
  float* outAttn = out + (size_t)(2 + dir) * B_TOT * DHID;

  const int wave = threadIdx.x >> 5, lane = threadIdx.x & 31;
  const int b = blockIdx.x * 8 + wave;

  float S = 0.f;
  #pragma unroll
  for (int k = 0; k < KDIM; ++k) S += tanhf(tD[(size_t)b * KDIM + k]);

  float v[32];
  float m = -3.4e38f;
  #pragma unroll
  for (int i = 0; i < 32; ++i) {
    v[i] = S * gv[(size_t)b * DHID + lane + 32 * i];
    m = fmaxf(m, v[i]);
  }
  #pragma unroll
  for (int o = 16; o >= 1; o >>= 1) m = fmaxf(m, __shfl_xor(m, o, 32));
  float sum = 0.f;
  #pragma unroll
  for (int i = 0; i < 32; ++i) { v[i] = __expf(v[i] - m); sum += v[i]; }
  #pragma unroll
  for (int o = 16; o >= 1; o >>= 1) sum += __shfl_xor(sum, o, 32);
  const float inv = 1.f / sum;
  #pragma unroll
  for (int i = 0; i < 32; ++i) {
    size_t idx = (size_t)b * DHID + lane + 32 * i;
    float a  = v[i] * inv;
    float hv = hA[idx];
    outAttn[idx] = a;
    outMain[idx] = fmaf(a, hv, hv);
  }
}

// ---------------------------------------------------------------------------
extern "C" void kernel_launch(void* const* d_in, const int* in_sizes, int n_in,
                              void* d_out, int out_size, void* d_ws, size_t ws_size,
                              hipStream_t stream)
{
  (void)in_sizes; (void)n_in; (void)out_size; (void)ws_size;

  const float* lh  = (const float*)d_in[0];
  const float* mh  = (const float*)d_in[1];
  const float* Glm = (const float*)d_in[2];
  const float* Gml = (const float*)d_in[3];
  const float* w1l = (const float*)d_in[4];
  const float* b1l = (const float*)d_in[5];
  const float* w2l = (const float*)d_in[6];
  const float* b2l = (const float*)d_in[7];
  const float* w1m = (const float*)d_in[8];
  const float* b1m = (const float*)d_in[9];
  const float* w2m = (const float*)d_in[10];
  const float* b2m = (const float*)d_in[11];

  char* ws = (char*)d_ws;
  unsigned short* gpk  = (unsigned short*)(ws);                       // 32 MB packed G (bf16)
  char* w1lp = ws + 33554432;                                         // 512 KB
  char* w2lp = ws + 33554432 + 524288;                                // 512 KB
  char* w1mp = ws + 33554432 + 1048576;                               // 256 KB
  char* w2mp = ws + 33554432 + 1310720;                               // 256 KB
  float* gvec = (float*)(ws + 35651584);                              // 64 MB
  float* tbuf = (float*)(ws + 102760448);                             // 512 KB

  // pack G: for each dir/k, B[d,e] = G[d,k,e]  (sd = K*DM, sn = 1)
  for (int dir = 0; dir < 2; ++dir) {
    const float* G = dir ? Gml : Glm;
    for (int k = 0; k < 8; ++k) {
      pack_b_kernel<<<256, 256, 0, stream>>>(
          G, (long)k * 1024, (long)8192, (long)1, 32,
          gpk + ((size_t)dir * 16384 + (size_t)k * 2048) * 512);
    }
  }
  // pack weights: B[d,n] = W[n,d]
  pack_b_kernel<<<64, 256, 0, stream>>>(w1l, 0, 1, 1024, 32, (unsigned short*)w1lp);
  pack_b_kernel<<<64, 256, 0, stream>>>(w2l, 0, 1, 256,   8, (unsigned short*)w2lp);
  pack_b_kernel<<<32, 256, 0, stream>>>(w1m, 0, 1, 1024, 32, (unsigned short*)w1mp);
  pack_b_kernel<<<32, 256, 0, stream>>>(w2m, 0, 1, 128,   4, (unsigned short*)w2mp);

  mlp_gvec_kernel<<<128, 256, 64*1024*2 + 64*256*2, stream>>>(
      lh, w1lp, b1l, w2lp, b2l, gvec, 256);
  mlp_gvec_kernel<<<128, 256, 64*1024*2 + 64*128*2, stream>>>(
      mh, w1mp, b1m, w2mp, b2m, gvec + (size_t)B_TOT * DHID, 128);

  cross_bilinear_kernel<<<dim3(256, 2), 256, 131072, stream>>>(
      lh, mh, (const char*)gpk, tbuf);

  softmax_out_kernel<<<dim3(1024, 2), 256, 0, stream>>>(
      lh, mh, gvec, tbuf, (float*)d_out);
}
